// NoisyGating_22436909154697
// MI455X (gfx1250) — matrix operand order repaired and checked by reading, not compiled, not run on hardware
//
#include <hip/hip_runtime.h>

#define TOKENS   8192
#define DDIM     4096
#define EXPERTS  64
#define NKB      (DDIM / 32)   // 128 k-steps of 32
#define NCHUNK   (NKB / 2)     // 64 chunks of 2 k-steps

typedef __attribute__((ext_vector_type(16))) __bf16        v16bf;
typedef __attribute__((ext_vector_type(2)))  __bf16        v2bf;
typedef __attribute__((ext_vector_type(8)))  float         v8f;
typedef __attribute__((ext_vector_type(8)))  unsigned int  v8u;
typedef __attribute__((ext_vector_type(4)))  float         f4v;

// ---- f32 pair -> packed bf16 dword (RNE) ----
#if __has_builtin(__builtin_amdgcn_cvt_pk_bf16_f32)
static __device__ __forceinline__ unsigned int pack_bf16(float lo, float hi) {
  v2bf p = __builtin_amdgcn_cvt_pk_bf16_f32(lo, hi);
  return __builtin_bit_cast(unsigned int, p);
}
#else
// fptrunc f32->bf16 is RNE; gfx1250 backend selects native cvt for pairs.
static __device__ __forceinline__ unsigned int pack_bf16(float lo, float hi) {
  v2bf p;
  p.x = (__bf16)lo;
  p.y = (__bf16)hi;
  return __builtin_bit_cast(unsigned int, p);
}
#endif

// Fragments per weight matrix: NKB * 4 expert tiles * 32 lanes * 8 dwords
#define FRAG_DWORDS_PER_MAT (NKB * 4 * 32 * 8) // 131072 dwords = 512 KB

// ---------------------------------------------------------------------------
// Pre-pass: pack w_gate / w_noise (f32, [D,E] row-major) into bf16 WMMA
// B-fragment layout: frag(kb,t,lane) holds column e = 16*t + (lane&15),
// K = kb*32 + (lane>>4)*8 + {0..7, 16..23}, two bf16 per dword (even K in [15:0]).
// ---------------------------------------------------------------------------
__global__ void pack_weights_kernel(const float* __restrict__ wg,
                                    const float* __restrict__ wn,
                                    unsigned int* __restrict__ out) {
  int tid = blockIdx.x * blockDim.x + threadIdx.x;
  const float* w = (tid < FRAG_DWORDS_PER_MAT) ? wg : wn;
  int id   = (tid < FRAG_DWORDS_PER_MAT) ? tid : tid - FRAG_DWORDS_PER_MAT;
  int v    = id & 7;          // dword within fragment
  int lane = (id >> 3) & 31;  // lane
  int t    = (id >> 8) & 3;   // expert tile
  int kb   = id >> 10;        // k-block (0..NKB-1)
  int e    = t * 16 + (lane & 15);
  int k0   = kb * 32 + (lane >> 4) * 8 + (v >> 2) * 16 + 2 * (v & 3);
  out[tid] = pack_bf16(w[(size_t)k0 * EXPERTS + e],
                       w[(size_t)(k0 + 1) * EXPERTS + e]);
}

// ---- A-fragment raw (4x float4, nontemporal: x is streamed exactly once) ----
struct ARaw { f4v q0, q1, q2, q3; };

static __device__ __forceinline__ void loadA(ARaw& a, const f4v* px) {
  a.q0 = __builtin_nontemporal_load(px + 0);  // K +0..3
  a.q1 = __builtin_nontemporal_load(px + 1);  // K +4..7
  a.q2 = __builtin_nontemporal_load(px + 4);  // K +16..19
  a.q3 = __builtin_nontemporal_load(px + 5);  // K +20..23
}

static __device__ __forceinline__ v16bf cvtA(const ARaw& a) {
  v8u ap;
  ap[0] = pack_bf16(a.q0.x, a.q0.y); ap[1] = pack_bf16(a.q0.z, a.q0.w);
  ap[2] = pack_bf16(a.q1.x, a.q1.y); ap[3] = pack_bf16(a.q1.z, a.q1.w);
  ap[4] = pack_bf16(a.q2.x, a.q2.y); ap[5] = pack_bf16(a.q2.z, a.q2.w);
  ap[6] = pack_bf16(a.q3.x, a.q3.y); ap[7] = pack_bf16(a.q3.z, a.q3.w);
  return __builtin_bit_cast(v16bf, ap);
}

#define WMMA_BF16(A, B, C) \
  __builtin_amdgcn_wmma_f32_16x16x32_bf16(false, (A), false, (B), (short)0, (C), false, false)

// One k-step: A from registers, B fragments read from LDS (ds_load_b128 x16).
static __device__ __forceinline__ void stepWMMA(const ARaw& ar,
                                                const v8u* bgL, const v8u* bnL,
                                                int lane,
                                                v8f (&accG)[4], v8f (&accN)[4]) {
  v16bf A = cvtA(ar);
  v16bf Bg[4], Bn[4];
#pragma unroll
  for (int t = 0; t < 4; ++t) {
    Bg[t] = __builtin_bit_cast(v16bf, bgL[t * 32 + lane]);
    Bn[t] = __builtin_bit_cast(v16bf, bnL[t * 32 + lane]);
  }
#pragma unroll
  for (int t = 0; t < 4; ++t) {
    accG[t] = WMMA_BF16(A, Bg[t], accG[t]);
    accN[t] = WMMA_BF16(A, Bn[t], accN[t]);
  }
}

// ---------------------------------------------------------------------------
// Fused noisy-gating kernel. 4 waves / WG; each wave: 16 tokens x 64 experts.
//  - x (HBM stream): per-wave register double-buffered, nontemporal loads.
//  - B fragments: staged through LDS, shared by all 4 waves (4x L2 traffic
//    cut), double-buffered in 2-kb chunks; stage-loads issued at chunk start,
//    ds_stores at chunk end so the chunk's WMMAs hide the global latency.
// ---------------------------------------------------------------------------
__global__ void __launch_bounds__(128)
noisy_gating_kernel(const float* __restrict__ x,
                    const float* __restrict__ b_gate,
                    const float* __restrict__ b_noise,
                    const float* __restrict__ noise,
                    const unsigned int* __restrict__ wfrag,
                    float* __restrict__ out) {
  const int tidx  = threadIdx.x;
  const int wave  = tidx >> 5;
  const int lane  = tidx & 31;
  const int r     = lane & 15;   // row-in-tile for A / column for B,C
  const int half  = lane >> 4;   // K-half for A/B, row-half for C
  const int mBase = blockIdx.x * 64 + wave * 16;

  // [buf][kbIn][mat][t*32+lane] : 32 KB
  __shared__ v8u   ldsB[2][2][2][128];
  __shared__ float ldsL[4][16][EXPERTS + 1]; // noisy logits, 16.6 KB

  v8f accG[4], accN[4];
#pragma unroll
  for (int t = 0; t < 4; ++t) { accG[t] = (v8f){0,0,0,0,0,0,0,0}; accN[t] = accG[t]; }

  // rolling pointers
  const f4v* pxs = (const f4v*)(x + (size_t)(mBase + r) * DDIM) + half * 2;
  const v8u* pgS = (const v8u*)wfrag + tidx;                         // gate frags
  const v8u* pnS = (const v8u*)(wfrag + FRAG_DWORDS_PER_MAT) + tidx; // noise frags

  ARaw arA[2], arB[2];
  v8u  s0, s1, s2, s3;

#define STAGE_LOAD() do { s0 = pgS[0]; s1 = pgS[128]; s2 = pnS[0]; s3 = pnS[128]; \
                          pgS += 256; pnS += 256; } while (0)
#define STAGE_STORE(buf) do { ldsB[buf][0][0][tidx] = s0; ldsB[buf][1][0][tidx] = s1; \
                              ldsB[buf][0][1][tidx] = s2; ldsB[buf][1][1][tidx] = s3; } while (0)
#define LOAD_A2(arr) do { loadA(arr[0], pxs); loadA(arr[1], pxs + 8); pxs += 16; } while (0)
#define COMPUTE(arr, buf) do { \
    stepWMMA(arr[0], &ldsB[buf][0][0][0], &ldsB[buf][0][1][0], lane, accG, accN); \
    stepWMMA(arr[1], &ldsB[buf][1][0][0], &ldsB[buf][1][1][0], lane, accG, accN); } while (0)

  // ---- prologue: stage chunk 0 into buf 0, A chunk 0 into arA ----
  STAGE_LOAD();
  STAGE_STORE(0);
  LOAD_A2(arA);
  __syncthreads();

  // ---- steady state: 31 iterations, chunk pair (c, c+1) ----
  for (int c = 0; c < NCHUNK - 2; c += 2) {
    STAGE_LOAD();            // chunk c+1 -> regs
    LOAD_A2(arB);            // A chunk c+1
    COMPUTE(arA, 0);         // chunk c from buf 0
    STAGE_STORE(1);
    __syncthreads();

    STAGE_LOAD();            // chunk c+2 -> regs
    LOAD_A2(arA);            // A chunk c+2
    COMPUTE(arB, 1);         // chunk c+1 from buf 1
    STAGE_STORE(0);
    __syncthreads();
  }
  // ---- tail: chunk 62 in arA/buf0; stage+consume chunk 63 ----
  STAGE_LOAD();
  LOAD_A2(arB);
  COMPUTE(arA, 0);
  STAGE_STORE(1);
  __syncthreads();
  COMPUTE(arB, 1);

#undef STAGE_LOAD
#undef STAGE_STORE
#undef LOAD_A2
#undef COMPUTE

  // ---- bias + softplus + noise, stage noisy logits in LDS ----
  // C/D layout: VGPR j -> row (j + 8*half), column e = 16*t + r
#pragma unroll
  for (int t = 0; t < 4; ++t) {
    const int   e  = t * 16 + r;
    const float bg = b_gate[e];
    const float bn = b_noise[e];
#pragma unroll
    for (int j = 0; j < 8; ++j) {
      const int   row   = j + half * 8;
      const int   token = mBase + row;
      const float g  = accG[t][j] + bg;
      const float z  = accN[t][j] + bn;
      const float sp = fmaxf(z, 0.f) + log1pf(__expf(-fabsf(z))); // stable softplus
      const float nz = noise[(size_t)token * EXPERTS + e];
      ldsL[wave][row][e] = g + nz * sp;
    }
  }
  __syncthreads();

  // ---- per-token top-2, softmax over the 2, one-hot mask ----
  if (lane < 16) {
    const int token = mBase + lane;
    float v0 = -3.0e38f, v1 = -3.0e38f;
    int   i0 = 0, i1 = 0;
#pragma unroll 8
    for (int e = 0; e < EXPERTS; ++e) {
      float v = ldsL[wave][lane][e];
      if (v > v0)      { v1 = v0; i1 = i0; v0 = v; i0 = e; }
      else if (v > v1) { v1 = v;  i1 = e; }
    }
    const float e1 = __expf(v1 - v0);
    const float s  = 1.f + e1;
    out[token * 2 + 0] = 1.f / s;
    out[token * 2 + 1] = e1 / s;
    float* oidx = out + (size_t)TOKENS * 2;
    oidx[token * 2 + 0] = (float)i0;
    oidx[token * 2 + 1] = (float)i1;
    float* om = out + (size_t)TOKENS * 4 + (size_t)token * EXPERTS;
#pragma unroll
    for (int e = 0; e < EXPERTS; ++e)
      om[e] = (e == i0 || e == i1) ? 1.f : 0.f;
  }
}

// ---------------------------------------------------------------------------
extern "C" void kernel_launch(void* const* d_in, const int* in_sizes, int n_in,
                              void* d_out, int out_size, void* d_ws, size_t ws_size,
                              hipStream_t stream) {
  const float* x       = (const float*)d_in[0];
  const float* w_gate  = (const float*)d_in[1];
  const float* b_gate  = (const float*)d_in[2];
  const float* w_noise = (const float*)d_in[3];
  const float* b_noise = (const float*)d_in[4];
  const float* noise   = (const float*)d_in[5];

  unsigned int* wfrag = (unsigned int*)d_ws; // needs 2 * 512 KB = 1 MB scratch

  pack_weights_kernel<<<(2 * FRAG_DWORDS_PER_MAT) / 256, 256, 0, stream>>>(
      w_gate, w_noise, wfrag);

  noisy_gating_kernel<<<TOKENS / 64, 128, 0, stream>>>(
      x, b_gate, b_noise, noise, wfrag, (float*)d_out);
}
